// MonotonicAttention_3375844295003
// MI455X (gfx1250) — matrix operand order repaired and checked by reading, not compiled, not run on hardware
//
#include <hip/hip_runtime.h>
#include <hip/hip_bf16.h>

typedef __attribute__((ext_vector_type(16))) _Float16 v16h;
typedef __attribute__((ext_vector_type(8)))  float    v8f;
typedef __attribute__((ext_vector_type(4)))  float    f32x4;
typedef __attribute__((ext_vector_type(4)))  int      i32x4;

#define B_DIM 64
#define L_DIM 1024
#define D_DIM 512
#define A_DIM 512
#define NEG_INF_F (-1e9f)

#if defined(__AMDGCN__) && __has_builtin(__builtin_amdgcn_tanhf)
#define HAVE_NATIVE_TANH 1
#else
#define HAVE_NATIVE_TANH 0
#endif

#if defined(__AMDGCN__) && \
    __has_builtin(__builtin_amdgcn_global_load_async_to_lds_b128) && \
    __has_builtin(__builtin_amdgcn_s_wait_asynccnt)
#define HAVE_ASYNC_LDS 1
typedef __attribute__((address_space(1))) i32x4 g_i32x4;   // global int4
typedef __attribute__((address_space(3))) i32x4 l_i32x4;   // LDS int4
#endif

__device__ __forceinline__ float fast_tanh(float x) {
#if HAVE_NATIVE_TANH
  return __builtin_amdgcn_tanhf(x);          // gfx1250 V_TANH_F32 (trans op)
#else
#warning "gfx1250 native tanh builtin NOT available; using exp-based fallback"
  float e = __expf(-2.0f * fabsf(x));
  float t = (1.0f - e) / (1.0f + e);
  return copysignf(t, x);
#endif
}

__device__ __forceinline__ float fast_sigmoid(float x) {
  return 0.5f * (1.0f + fast_tanh(0.5f * x));
}

// ---------------------------------------------------------------------------
// Kernel 0: weight-normalized v:  v_eff = g * v / ||v||,  scal = v_bias + r
// ---------------------------------------------------------------------------
__global__ void __launch_bounds__(A_DIM) veff_kernel(
    const float* __restrict__ vw, const float* __restrict__ vg,
    const float* __restrict__ vbias, const float* __restrict__ r,
    float* __restrict__ veff, float* __restrict__ scal) {
  __shared__ float s[A_DIM];
  const int tid = threadIdx.x;
  float w = vw[tid];
  s[tid] = w * w;
  __syncthreads();
  for (int off = A_DIM / 2; off > 0; off >>= 1) {
    if (tid < off) s[tid] += s[tid + off];
    __syncthreads();
  }
  veff[tid] = vg[0] * w * rsqrtf(s[0]);
  if (tid == 0) scal[0] = vbias[0] + r[0];
}

// ---------------------------------------------------------------------------
// Kernel 1: qb[B,A] = decoder_h[B,D] @ W[A,D]^T + b   via v_wmma_f32_16x16x32_f16
// One wave per 16x16 output tile; 16 k-steps of K=32.
// ---------------------------------------------------------------------------
__global__ void __launch_bounds__(32) qb_wmma_kernel(
    const float* __restrict__ dh, const float* __restrict__ W,
    const float* __restrict__ bias, float* __restrict__ qb) {
  const int tile = blockIdx.x;            // 128 tiles: 4 (M) x 32 (N)
  const int m0 = (tile >> 5) * 16;
  const int n0 = (tile & 31) * 16;
  const int lane = threadIdx.x;
  const int grp  = lane >> 4;             // half-wave id
  const int mr = m0 + (lane & 15);        // A row (batch)
  const int nr = n0 + (lane & 15);        // B col (A-dim)

  v8f c = {};
  for (int kk = 0; kk < D_DIM; kk += 32) {
    v16h a, bm;
#pragma unroll
    for (int h = 0; h < 8; ++h) {
      a[h]     = (_Float16)dh[mr * D_DIM + kk + grp * 8 + h];
      a[h + 8] = (_Float16)dh[mr * D_DIM + kk + 16 + grp * 8 + h];
    }
#pragma unroll
    for (int h = 0; h < 16; ++h) {
      bm[h] = (_Float16)W[nr * D_DIM + kk + (h < 8 ? grp * 8 + h : 16 + grp * 8 + (h - 8))];   // B[k][n] = W[n][k]
    }
    c = __builtin_amdgcn_wmma_f32_16x16x32_f16(
        false, a, false, bm, (short)0, c, false, false);
  }
#pragma unroll
  for (int rr = 0; rr < 8; ++rr) {
    int m = m0 + rr + grp * 8;
    int n = n0 + (lane & 15);
    qb[m * A_DIM + n] = c[rr] + bias[n];
  }
}

// ---------------------------------------------------------------------------
// Kernel 2: the bandwidth-bound hot loop — streams key[B,L,A] (134 MB, read
// once -> non-temporal b128 loads). One wave per two l positions (8 loads in
// flight), qb row + veff staged in LDS via async-to-LDS when available.
// ---------------------------------------------------------------------------
__global__ void __launch_bounds__(256) energy_p_kernel(
    const float* __restrict__ key, const float* __restrict__ noise,
    const float* __restrict__ mask, const float* __restrict__ qb,
    const float* __restrict__ veff, const float* __restrict__ scal,
    float* __restrict__ psel) {
  __shared__ alignas(16) float s_q[A_DIM];
  __shared__ alignas(16) float s_v[A_DIM];
  const int blocks_per_b = L_DIM / 16;     // 64
  const int b  = blockIdx.x / blocks_per_b;
  const int l0 = (blockIdx.x % blocks_per_b) * 16;
  const int tid = threadIdx.x;

#if HAVE_ASYNC_LDS
  // 256 threads x one b128 each covers s_q (2KB) and s_v (2KB)
  if (tid < 128) {
    __builtin_amdgcn_global_load_async_to_lds_b128(
        (g_i32x4*)(unsigned long long)(qb + b * A_DIM + tid * 4),
        (l_i32x4*)&s_q[tid * 4], 0, 0);
  } else {
    int t = tid - 128;
    __builtin_amdgcn_global_load_async_to_lds_b128(
        (g_i32x4*)(unsigned long long)(veff + t * 4),
        (l_i32x4*)&s_v[t * 4], 0, 0);
  }
  __builtin_amdgcn_s_wait_asynccnt(0);
#else
  s_q[tid]       = qb[b * A_DIM + tid];
  s_q[tid + 256] = qb[b * A_DIM + tid + 256];
  s_v[tid]       = veff[tid];
  s_v[tid + 256] = veff[tid + 256];
#endif
  __syncthreads();

  const int wave = tid >> 5;
  const int lane = tid & 31;
  const int la = l0 + wave;        // first l
  const int lb = la + 8;           // second l
  const f32x4* ka = (const f32x4*)(key + ((size_t)b * L_DIM + la) * A_DIM);
  const f32x4* kb = (const f32x4*)(key + ((size_t)b * L_DIM + lb) * A_DIM);

  f32x4 ra[4], rb[4];
#pragma unroll
  for (int i = 0; i < 4; ++i) {    // 8 non-temporal b128 loads in flight
    ra[i] = __builtin_nontemporal_load(&ka[i * 32 + lane]);
    rb[i] = __builtin_nontemporal_load(&kb[i * 32 + lane]);
  }

  float acc_a = 0.0f, acc_b = 0.0f;
#pragma unroll
  for (int i = 0; i < 4; ++i) {
    int a = (i * 32 + lane) * 4;
    float q0 = s_q[a + 0], q1 = s_q[a + 1], q2 = s_q[a + 2], q3 = s_q[a + 3];
    float v0 = s_v[a + 0], v1 = s_v[a + 1], v2 = s_v[a + 2], v3 = s_v[a + 3];
    acc_a += fast_tanh(q0 + ra[i].x) * v0;
    acc_a += fast_tanh(q1 + ra[i].y) * v1;
    acc_a += fast_tanh(q2 + ra[i].z) * v2;
    acc_a += fast_tanh(q3 + ra[i].w) * v3;
    acc_b += fast_tanh(q0 + rb[i].x) * v0;
    acc_b += fast_tanh(q1 + rb[i].y) * v1;
    acc_b += fast_tanh(q2 + rb[i].z) * v2;
    acc_b += fast_tanh(q3 + rb[i].w) * v3;
  }
#pragma unroll
  for (int off = 16; off > 0; off >>= 1) {
    acc_a += __shfl_xor(acc_a, off, 32);
    acc_b += __shfl_xor(acc_b, off, 32);
  }

  if (lane == 0) {
    float sc = scal[0];
#pragma unroll
    for (int s = 0; s < 2; ++s) {
      int l = (s == 0) ? la : lb;
      float acc = (s == 0) ? acc_a : acc_b;
      size_t idx = (size_t)b * L_DIM + l;
      float mj = mask[idx];
      float mn = (l + 1 < L_DIM) ? mask[idx + 1] : 0.0f;
      float endm = mj * (1.0f - mn);
      float e = (mj > 0.0f) ? (acc + sc) : NEG_INF_F;
      float p = fast_sigmoid(e + noise[idx]);
      psel[idx] = (endm > 0.0f) ? endm : p;
    }
  }
}

// ---------------------------------------------------------------------------
// Kernel 3: per-row linear recurrence x_j = a_j*x_{j-1} + b_j via two-level
// inclusive scan with pair combine (a1,b1)∘(a2,b2) = (a1*a2, a2*b1 + b2).
// ---------------------------------------------------------------------------
__global__ void __launch_bounds__(1024) scan_kernel(
    const float* __restrict__ psel, const float* __restrict__ prev_att,
    float* __restrict__ alpha) {
  __shared__ float sA[32];
  __shared__ float sB[32];
  const int b = blockIdx.x;
  const int l = threadIdx.x;
  const size_t idx = (size_t)b * L_DIM + l;

  float p  = psel[idx];
  float av = (l == 0) ? 1.0f : (1.0f - psel[idx - 1]);   // shifted (1 - p_{j-1})
  float bv = prev_att[idx];
  const int lane = l & 31;
  const int wave = l >> 5;

#pragma unroll
  for (int off = 1; off < 32; off <<= 1) {               // wave inclusive scan
    float pa = __shfl_up(av, off, 32);
    float pb = __shfl_up(bv, off, 32);
    if (lane >= off) { bv = av * pb + bv; av = pa * av; }
  }
  if (lane == 31) { sA[wave] = av; sB[wave] = bv; }
  __syncthreads();
  if (wave == 0) {                                       // scan 32 aggregates
    float wa = sA[lane], wb = sB[lane];
#pragma unroll
    for (int off = 1; off < 32; off <<= 1) {
      float pa = __shfl_up(wa, off, 32);
      float pb = __shfl_up(wb, off, 32);
      if (lane >= off) { wb = wa * pb + wb; wa = pa * wa; }
    }
    sA[lane] = wa; sB[lane] = wb;
  }
  __syncthreads();
  float x = (wave == 0) ? bv : (av * sB[wave - 1] + bv);
  alpha[idx] = p * x;
}

// ---------------------------------------------------------------------------
extern "C" void kernel_launch(void* const* d_in, const int* in_sizes, int n_in,
                              void* d_out, int out_size, void* d_ws, size_t ws_size,
                              hipStream_t stream) {
  const float* dh    = (const float*)d_in[0];   // decoder_h [B,D]
  const float* key   = (const float*)d_in[1];   // key [B,L,A]
  // d_in[2] = encoder_outputs: unused by the reference computation
  const float* patt  = (const float*)d_in[3];   // prev_att [B,L]
  const float* noise = (const float*)d_in[4];   // noise [B,L]
  const float* mask  = (const float*)d_in[5];   // mask [B,L]
  const float* W     = (const float*)d_in[6];   // W [A,D]
  const float* bias  = (const float*)d_in[7];   // b [A]
  const float* vw    = (const float*)d_in[8];   // v_weight [1,A]
  const float* vg    = (const float*)d_in[9];   // v_g [1]
  const float* vb    = (const float*)d_in[10];  // v_bias [1]
  const float* r     = (const float*)d_in[11];  // r [1]
  float* out = (float*)d_out;

  float* ws      = (float*)d_ws;
  float* ws_veff = ws;                           // 512
  float* ws_scal = ws + A_DIM;                   // 1 (padded to 16)
  float* ws_q    = ws + A_DIM + 16;              // B*A = 32768 (16B aligned)
  float* ws_p    = ws_q + B_DIM * A_DIM;         // B*L = 65536

  veff_kernel<<<1, A_DIM, 0, stream>>>(vw, vg, vb, r, ws_veff, ws_scal);
  qb_wmma_kernel<<<(B_DIM / 16) * (A_DIM / 16), 32, 0, stream>>>(dh, W, bias, ws_q);
  energy_p_kernel<<<B_DIM * (L_DIM / 16), 256, 0, stream>>>(
      key, noise, mask, ws_q, ws_veff, ws_scal, ws_p);
  scan_kernel<<<B_DIM, 1024, 0, stream>>>(ws_p, patt, out);
}
